// CausalSelfAttention_82806969467406
// MI455X (gfx1250) — matrix operand order repaired and checked by reading, not compiled, not run on hardware
//
#include <hip/hip_runtime.h>
#include <hip/hip_bf16.h>
#include <math.h>

// ---------------------------------------------------------------------------
// Types for CDNA5 WMMA (wave32): v_wmma_f32_16x16x32_bf16
// ---------------------------------------------------------------------------
typedef __bf16 bf16;
typedef __attribute__((ext_vector_type(16))) __bf16 bf16x16;
typedef __attribute__((ext_vector_type(8)))  __bf16 bf16x8;
typedef __attribute__((ext_vector_type(8)))  float  f32x8;
typedef __attribute__((ext_vector_type(4)))  float  f32x4;

static __device__ __forceinline__ bf16 f2bf(float f) {
  union { float f; unsigned u; } a; a.f = f;
  unsigned r = a.u + 0x7FFFu + ((a.u >> 16) & 1u);   // round-to-nearest-even
  union { unsigned short s; bf16 h; } b; b.s = (unsigned short)(r >> 16);
  return b.h;
}

static __device__ __forceinline__ bf16x16 join8(bf16x8 lo, bf16x8 hi) {
  bf16x16 v;
#pragma unroll
  for (int i = 0; i < 8; ++i) { v[i] = lo[i]; v[8 + i] = hi[i]; }
  return v;
}

// A fragment (16x32 bf16, row-major source): lane L -> row M = L%16,
// elements e: K = ks + (L>=16?16:0) + e  => two 16B loads.
static __device__ __forceinline__ bf16x16 load_a_frag(const bf16* __restrict__ A,
                                                      int lane, int row, int ldk, int ks) {
  int kb = (lane >= 16) ? 16 : 0;
  const bf16* p = A + row * ldk + ks + kb;
  bf16x8 lo = *(const bf16x8*)p;
  bf16x8 hi = *(const bf16x8*)(p + 8);
  return join8(lo, hi);
}

// B fragment (32x16 bf16) from a K-major (transposed) source: lane L -> col
// N = L%16, elements e: K = ks + (L>=16?16:0) + e  => 16 contiguous bf16.
static __device__ __forceinline__ bf16x16 load_b_frag(const bf16* __restrict__ Bt,
                                                      int lane, int col, int ldk, int ks) {
  int kb = (lane >= 16) ? 16 : 0;
  const bf16* p = Bt + col * ldk + ks + kb;
  bf16x8 lo = *(const bf16x8*)p;
  bf16x8 hi = *(const bf16x8*)(p + 8);
  return join8(lo, hi);
}

// pipelined-load helper (pre-biased pointer; ks in elements)
static __device__ __forceinline__ bf16x16 ldA(const bf16* __restrict__ ap, int ks) {
  return join8(*(const bf16x8*)(ap + ks), *(const bf16x8*)(ap + ks + 8));
}

// ---------------------------------------------------------------------------
// 32x64-strip GEMM core over K=768, fully unrolled. Two M-tiles share every
// B fragment: 12 b128 loads feed 8 WMMAs per K-step (1.5 loads/WMMA), and
// each B fragment has two independent consumers so the scheduler can cover
// load latency with the second M-tile's WMMA on the previous fragment.
// ---------------------------------------------------------------------------
static __device__ __forceinline__ void gemm_strip2_768(const bf16* __restrict__ A,
                                                       const bf16* __restrict__ Bt,
                                                       int lane, int arow0, int n0g,
                                                       f32x8 acc0[4], f32x8 acc1[4]) {
  constexpr int K = 768, NK = K / 32;          // 24 K-steps
  const int kbA = (lane >= 16) ? 16 : 0;
  const int kbB = (lane >= 16) ? 16 : 0;
  const bf16* ap0 = A + arow0 * K + kbA;       // M-tile 0 (rows m0..m0+15)
  const bf16* ap1 = ap0 + 16 * K;              // M-tile 1 (rows m0+16..m0+31)
  const bf16* bp[4];
#pragma unroll
  for (int s2 = 0; s2 < 4; ++s2)
    bp[s2] = Bt + (n0g + s2 * 16 + (lane & 15)) * K + kbB;

#pragma unroll
  for (int it = 0; it < NK; ++it) {
    const int ks = it * 32;
    bf16x16 a0 = ldA(ap0, ks);
    bf16x16 a1 = ldA(ap1, ks);
#pragma unroll
    for (int s2 = 0; s2 < 4; ++s2) {
      bf16x16 b = join8(*(const bf16x8*)(bp[s2] + ks), *(const bf16x8*)(bp[s2] + ks + 8));
      acc0[s2] = __builtin_amdgcn_wmma_f32_16x16x32_bf16(false, a0, false, b,
                                                         (short)0, acc0[s2], false, false);
      acc1[s2] = __builtin_amdgcn_wmma_f32_16x16x32_bf16(false, a1, false, b,
                                                         (short)0, acc1[s2], false, false);
    }
  }
}

// ---------------------------------------------------------------------------
// Stage 0: dtype conversion / weight transpose
// ---------------------------------------------------------------------------
__global__ void cvt_f32_bf16(const float* __restrict__ in, bf16* __restrict__ out, int n) {
  int i = blockIdx.x * blockDim.x + threadIdx.x;
  int stride = gridDim.x * blockDim.x;
  for (; i < n; i += stride) out[i] = f2bf(in[i]);
}

// out[n*K + k] = in[k*N + n]   (weights: [K,N] fp32 -> [N,K] bf16)
__global__ void transpose_cvt(const float* __restrict__ in, bf16* __restrict__ out,
                              int K, int N) {
  int i = blockIdx.x * blockDim.x + threadIdx.x;
  int total = K * N;
  int stride = gridDim.x * blockDim.x;
  for (; i < total; i += stride) {
    int n = i / K, k = i - n * K;
    out[i] = f2bf(in[k * N + n]);
  }
}

// ---------------------------------------------------------------------------
// Stage 1: QKV projection. One wave -> 32x64 output strip. Writes Q (scaled
// by 1/sqrt(64)) and K as [B,H,S,64] bf16, V transposed as [B,H,64,S] bf16.
// ---------------------------------------------------------------------------
__global__ __launch_bounds__(256) void qkv_gemm(const bf16* __restrict__ xbf,
                                                const bf16* __restrict__ WtIn,
                                                const float* __restrict__ b_in,
                                                bf16* __restrict__ Qb,
                                                bf16* __restrict__ Kb,
                                                bf16* __restrict__ Vt) {
  const int lane = threadIdx.x & 31, wave = threadIdx.x >> 5;
  int job = blockIdx.x * 8 + wave;          // 256 * 36 jobs
  int mt = job / 36, ng = job - mt * 36;
  int m0 = mt * 32, n0g = ng * 64;

  f32x8 acc[2][4] = {};
  gemm_strip2_768(xbf, WtIn, lane, m0 + (lane & 15), n0g, acc[0], acc[1]);

#pragma unroll
  for (int mh = 0; mh < 2; ++mh) {
#pragma unroll
    for (int s2 = 0; s2 < 4; ++s2) {
      int n = n0g + s2 * 16 + (lane & 15);
      float bias = b_in[n];
#pragma unroll
      for (int r = 0; r < 8; ++r) {
        int m = m0 + mh * 16 + r + ((lane >= 16) ? 8 : 0);
        float v = acc[mh][s2][r] + bias;
        int bb = m >> 12, sIdx = m & 4095;
        if (n < 768) {                       // Q (pre-scaled by 1/sqrt(d_head))
          int h = n >> 6, d = n & 63;
          Qb[(((bb * 12 + h) * 4096 + sIdx) << 6) + d] = f2bf(v * 0.125f);
        } else if (n < 1536) {               // K
          int c = n - 768; int h = c >> 6, d = c & 63;
          Kb[(((bb * 12 + h) * 4096 + sIdx) << 6) + d] = f2bf(v);
        } else {                             // V, transposed [B,H,64,S]
          int c = n - 1536; int h = c >> 6, d = c & 63;
          Vt[(((bb * 12 + h) * 64 + d) << 12) + sIdx] = f2bf(v);
        }
      }
    }
  }
}

// ---------------------------------------------------------------------------
// Stage 2: causal flash attention. One wave per 16-query tile; streaming
// softmax through wave-private LDS; WMMA for both Q·K^T and P·V. V-fragment
// global loads are issued BEFORE the LDS softmax phase so their latency
// hides under the ds round-trip and v_exp_f32 work.
// ---------------------------------------------------------------------------
__global__ __launch_bounds__(256) void attn(const bf16* __restrict__ Qb,
                                            const bf16* __restrict__ Kb,
                                            const bf16* __restrict__ Vt,
                                            bf16* __restrict__ Y) {
  __shared__ float Sst[8][512];   // 16x32 f32 scores per wave
  __shared__ bf16  Pst[8][512];   // 16x32 bf16 probs per wave
  __shared__ float Asc[8][16];    // per-row alpha
  __shared__ float Lrc[8][16];    // per-row 1/l

  const int lane = threadIdx.x & 31, wave = threadIdx.x >> 5;
  int job = blockIdx.x * 8 + wave;          // 24 * 256 jobs
  int bh = job >> 8;
  int qt = job & 255;
  int q0 = qt * 16;

  const bf16* Qrow = Qb + (bh * 4096) * 64;
  const bf16* Krow = Kb + (bh * 4096) * 64;
  const bf16* Vrow = Vt + (bh * 64) * 4096;

  // Q fragments for the whole tile (16 x 64 = two K-steps of 32)
  bf16x16 aQ0 = load_a_frag(Qrow, lane, q0 + (lane & 15), 64, 0);
  bf16x16 aQ1 = load_a_frag(Qrow, lane, q0 + (lane & 15), 64, 32);

  f32x8 O[4] = {};
  float m_i = -1e30f, l_i = 0.f;

  const int kmax = q0 + 16;                  // causal bound (exclusive)
  for (int kc = 0; kc < kmax; kc += 32) {
    if (kc + 32 < kmax)
      __builtin_prefetch(Krow + (kc + 32 + (lane & 15)) * 64, 0, 3);

    // ---- S = Q K^T for 32 keys (two 16-key subtiles) ----
    f32x8 sfr[2];
#pragma unroll
    for (int t = 0; t < 2; ++t) {
      int keyRow = kc + t * 16 + (lane & 15);
      bf16x16 b0 = load_b_frag(Krow, lane, keyRow, 64, 0);
      bf16x16 b1 = load_b_frag(Krow, lane, keyRow, 64, 32);
      f32x8 z = {};
      z = __builtin_amdgcn_wmma_f32_16x16x32_bf16(false, aQ0, false, b0, (short)0, z, false, false);
      z = __builtin_amdgcn_wmma_f32_16x16x32_bf16(false, aQ1, false, b1, (short)0, z, false, false);
      sfr[t] = z;
    }

    // ---- issue V-fragment loads NOW (independent of softmax) ----
    const int kb2 = (lane >= 16) ? 16 : 0;
    bf16x16 bV[4];
#pragma unroll
    for (int s2 = 0; s2 < 4; ++s2) {
      const bf16* vp = Vrow + (s2 * 16 + (lane & 15)) * 4096 + kc + kb2;
      bV[s2] = join8(*(const bf16x8*)vp, *(const bf16x8*)(vp + 8));
    }

    // ---- stage scores to LDS (C-frag layout -> row major) ----
#pragma unroll
    for (int t = 0; t < 2; ++t) {
      int colj = (lane & 15) + 16 * t;
#pragma unroll
      for (int r = 0; r < 8; ++r) {
        int row = r + ((lane >= 16) ? 8 : 0);
        Sst[wave][row * 32 + colj] = sfr[t][r];
      }
    }
    asm volatile("s_wait_dscnt 0" ::: "memory");

    // ---- streaming softmax: lanes 0..15 each own one query row ----
    if (lane < 16) {
      int q = q0 + lane;
      float vals[32];
#pragma unroll
      for (int c4 = 0; c4 < 8; ++c4) {
        f32x4 v = *(const f32x4*)&Sst[wave][lane * 32 + c4 * 4];
#pragma unroll
        for (int j = 0; j < 4; ++j) {
          int key = kc + c4 * 4 + j;
          vals[c4 * 4 + j] = (key <= q) ? v[j] : -1e30f;   // causal mask
        }
      }
      float mnew = m_i;
#pragma unroll
      for (int j = 0; j < 32; ++j) mnew = fmaxf(mnew, vals[j]);
      float alpha = __expf(m_i - mnew);
      float rowsum = 0.f;
#pragma unroll
      for (int j = 0; j < 32; ++j) {
        float p = __expf(vals[j] - mnew);
        rowsum += p;
        Pst[wave][lane * 32 + j] = f2bf(p);
      }
      l_i = l_i * alpha + rowsum;
      m_i = mnew;
      Asc[wave][lane] = alpha;
    }
    asm volatile("s_wait_dscnt 0" ::: "memory");

    // ---- rescale running output by alpha (per query row) ----
    float al[8];
#pragma unroll
    for (int r = 0; r < 8; ++r) al[r] = Asc[wave][r + ((lane >= 16) ? 8 : 0)];
#pragma unroll
    for (int s2 = 0; s2 < 4; ++s2)
#pragma unroll
      for (int r = 0; r < 8; ++r) O[s2][r] *= al[r];

    // ---- O += P · V  (A-frag of P from LDS, V frags already in registers) ----
    {
      int kb = (lane >= 16) ? 16 : 0;
      const bf16* pp = &Pst[wave][(lane & 15) * 32 + kb];
      bf16x8 plo = *(const bf16x8*)pp;
      bf16x8 phi = *(const bf16x8*)(pp + 8);
      bf16x16 aP = join8(plo, phi);
#pragma unroll
      for (int s2 = 0; s2 < 4; ++s2)
        O[s2] = __builtin_amdgcn_wmma_f32_16x16x32_bf16(false, aP, false, bV[s2],
                                                        (short)0, O[s2], false, false);
    }
  }

  // ---- finalize: divide by l, write Y bf16 [B*S, 768] ----
  if (lane < 16) Lrc[wave][lane] = 1.0f / l_i;
  asm volatile("s_wait_dscnt 0" ::: "memory");
  float li[8];
#pragma unroll
  for (int r = 0; r < 8; ++r) li[r] = Lrc[wave][r + ((lane >= 16) ? 8 : 0)];

  int bb = bh / 12, h = bh - bb * 12;
#pragma unroll
  for (int s2 = 0; s2 < 4; ++s2) {
    int d = s2 * 16 + (lane & 15);
#pragma unroll
    for (int r = 0; r < 8; ++r) {
      int q = q0 + r + ((lane >= 16) ? 8 : 0);
      int row = bb * 4096 + q;
      Y[row * 768 + h * 64 + d] = f2bf(O[s2][r] * li[r]);
    }
  }
}

// ---------------------------------------------------------------------------
// Stage 3: output projection -> fp32 d_out
// ---------------------------------------------------------------------------
__global__ __launch_bounds__(256) void out_gemm(const bf16* __restrict__ Ybf,
                                                const bf16* __restrict__ WtOut,
                                                const float* __restrict__ b_out,
                                                float* __restrict__ out) {
  const int lane = threadIdx.x & 31, wave = threadIdx.x >> 5;
  int job = blockIdx.x * 8 + wave;          // 256 * 12 jobs
  int mt = job / 12, ng = job - mt * 12;
  int m0 = mt * 32, n0g = ng * 64;

  f32x8 acc[2][4] = {};
  gemm_strip2_768(Ybf, WtOut, lane, m0 + (lane & 15), n0g, acc[0], acc[1]);

#pragma unroll
  for (int mh = 0; mh < 2; ++mh) {
#pragma unroll
    for (int s2 = 0; s2 < 4; ++s2) {
      int n = n0g + s2 * 16 + (lane & 15);
      float bias = b_out[n];
#pragma unroll
      for (int r = 0; r < 8; ++r) {
        int m = m0 + mh * 16 + r + ((lane >= 16) ? 8 : 0);
        out[m * 768 + n] = acc[mh][s2][r] + bias;
      }
    }
  }
}

// ---------------------------------------------------------------------------
// Launcher
// ---------------------------------------------------------------------------
extern "C" void kernel_launch(void* const* d_in, const int* in_sizes, int n_in,
                              void* d_out, int out_size, void* d_ws, size_t ws_size,
                              hipStream_t stream) {
  const float* x     = (const float*)d_in[0];   // [2,4096,768]
  const float* W_in  = (const float*)d_in[1];   // [768,2304]
  const float* b_in  = (const float*)d_in[2];   // [2304]
  const float* W_out = (const float*)d_in[3];   // [768,768]
  const float* b_out = (const float*)d_in[4];   // [768]
  float* out = (float*)d_out;                    // [2,4096,768] fp32

  char* ws = (char*)d_ws;
  bf16* xbf   = (bf16*)(ws + 0);          // 8192x768      bf16
  bf16* WtIn  = (bf16*)(ws + 12582912);   // 2304x768      bf16 (transposed)
  bf16* WtOut = (bf16*)(ws + 16121856);   // 768x768       bf16 (transposed)
  bf16* Qb    = (bf16*)(ws + 17301504);   // [2,12,4096,64] bf16 (pre-scaled)
  bf16* Kb    = (bf16*)(ws + 29884416);   // [2,12,4096,64] bf16
  bf16* Vt    = (bf16*)(ws + 42467328);   // [2,12,64,4096] bf16
  bf16* Ybf   = (bf16*)(ws + 55050240);   // 8192x768      bf16

  cvt_f32_bf16 <<<1024, 256, 0, stream>>>(x, xbf, 8192 * 768);
  transpose_cvt<<<1024, 256, 0, stream>>>(W_in,  WtIn,  768, 2304);
  transpose_cvt<<<512,  256, 0, stream>>>(W_out, WtOut, 768, 768);
  qkv_gemm     <<<1152, 256, 0, stream>>>(xbf, WtIn, b_in, Qb, Kb, Vt);
  attn         <<<768,  256, 0, stream>>>(Qb, Kb, Vt, Ybf);
  out_gemm     <<<384,  256, 0, stream>>>(Ybf, WtOut, b_out, out);
}
